// MultiMlpNet_75136157877067
// MI455X (gfx1250) — matrix operand order repaired and checked by reading, not compiled, not run on hardware
//
#include <hip/hip_runtime.h>

// Problem constants (match reference)
#define B_    4096
#define IN_   1024
#define V_    48
#define H_    512
#define OUT_  64
#define BT    64            // batch rows per block
#define NTHR  512           // 16 wave32 waves

typedef __attribute__((ext_vector_type(16))) __bf16 v16bf;
typedef __attribute__((ext_vector_type(8)))  __bf16 bf16x8;
typedef __attribute__((ext_vector_type(4)))  __bf16 bf16x4;
typedef __attribute__((ext_vector_type(2)))  __bf16 bf16x2;
typedef __attribute__((ext_vector_type(8)))  float  v8f;
typedef __attribute__((ext_vector_type(4)))  float  f32x4;

// ---- LDS layout (bf16 elements), double-buffered staging ------------------
// sX[2]  : x chunk       [64][72]
// sW[2]  : weight chunk  [512][72] (L1/L2), or [64][520] for L3 (fits in buf0)
// sH1    : h1            [64][520]
// sH2    : h2            [64][520]
#define SX_STR   72
#define SX_BUF   (64 * SX_STR)                 // 4608
#define SW_STR   72
#define SW3_STR  520
#define SH_STR   520
#define SW_BASE  (2 * SX_BUF)                  // 9216
#define SW_BUF   (512 * SW_STR)                // 36864
#define SH1_OFF  (SW_BASE + 2 * SW_BUF)        // 82944
#define SH2_OFF  (SH1_OFF + 64 * SH_STR)       // 116224
#define LDS_N    (SH2_OFF + 64 * SH_STR)       // 149504 bf16 = 299008 B

__device__ __forceinline__ v16bf ld_frag(const __bf16* p0, const __bf16* p1) {
    bf16x8 a = *(const bf16x8*)p0;
    bf16x8 b = *(const bf16x8*)p1;
    return __builtin_shufflevector(a, b, 0,1,2,3,4,5,6,7,8,9,10,11,12,13,14,15);
}

// ---- staging helpers: weight chunk is 64(k) x 512(n) f32, stored transposed
// [n][k] bf16.  Each thread owns 8 units of (2 adjacent k) x (4 adjacent n).
__device__ __forceinline__ void load_w_chunk(const float* __restrict__ wsrc,
                                             int tid, f32x4* a, f32x4* b) {
    #pragma unroll
    for (int it = 0; it < 8; ++it) {
        int u = it * NTHR + tid;
        int n = (u & 127) * 4;
        int k = (u >> 7) * 2;
        a[it] = *(const f32x4*)&wsrc[(size_t)k * H_ + n];
        b[it] = *(const f32x4*)&wsrc[(size_t)(k + 1) * H_ + n];
    }
}
__device__ __forceinline__ void cvt_store_w(__bf16* swb, int tid,
                                            const f32x4* a, const f32x4* b) {
    #pragma unroll
    for (int it = 0; it < 8; ++it) {
        int u = it * NTHR + tid;
        int n = (u & 127) * 4;
        int k = (u >> 7) * 2;
        #pragma unroll
        for (int j = 0; j < 4; ++j) {
            bf16x2 p;
            p.x = (__bf16)a[it][j];
            p.y = (__bf16)b[it][j];
            *(bf16x2*)&swb[(n + j) * SW_STR + k] = p;   // 4B-aligned ds_store_b32
        }
    }
}
__device__ __forceinline__ void load_x_chunk(const float* __restrict__ x,
                                             int row0, int kc, int tid, f32x4* xr) {
    #pragma unroll
    for (int it = 0; it < 2; ++it) {
        int e4 = (it * NTHR + tid) * 4;
        int m  = e4 >> 6;
        int kk = e4 & 63;
        xr[it] = *(const f32x4*)&x[(size_t)(row0 + m) * IN_ + kc * 64 + kk];
    }
}
__device__ __forceinline__ void cvt_store_x(__bf16* sxb, int tid, const f32x4* xr) {
    #pragma unroll
    for (int it = 0; it < 2; ++it) {
        int e4 = (it * NTHR + tid) * 4;
        int m  = e4 >> 6;
        int kk = e4 & 63;
        bf16x4 h;
        h.x = (__bf16)xr[it].x; h.y = (__bf16)xr[it].y;
        h.z = (__bf16)xr[it].z; h.w = (__bf16)xr[it].w;
        *(bf16x4*)&sxb[m * SX_STR + kk] = h;            // 8B-aligned ds_store_b64
    }
}

// one K=32 step of the 2x4-tile wave GEMM
__device__ __forceinline__ void mma_step(const __bf16* abase, int astr, int akb,
                                         const __bf16* bbuf, int bkb,
                                         int mb, int nb, int lo, int hi,
                                         v8f acc[2][4]) {
    v16bf A[2];
    #pragma unroll
    for (int i = 0; i < 2; ++i) {
        const __bf16* ar = abase + (mb + i * 16 + lo) * astr + akb + hi * 8;
        A[i] = ld_frag(ar, ar + 16);
    }
    #pragma unroll
    for (int j = 0; j < 4; ++j) {
        const __bf16* br = bbuf + (nb + j * 16 + lo) * SW_STR + bkb + hi * 8;
        v16bf Bv = ld_frag(br, br + 16);
        #pragma unroll
        for (int i = 0; i < 2; ++i) {
            acc[i][j] = __builtin_amdgcn_wmma_f32_16x16x32_bf16(
                false, A[i], false, Bv, (short)0, acc[i][j], false, false);
        }
    }
}

__global__ __launch_bounds__(NTHR, 1)
void mlp48_fused_kernel(const float* __restrict__ x,
                        const float* __restrict__ W1, const float* __restrict__ b1,
                        const float* __restrict__ W2, const float* __restrict__ b2,
                        const float* __restrict__ W3,
                        float* __restrict__ out)
{
    __shared__ __align__(16) __bf16 lds[LDS_N];

    const int tid  = threadIdx.x;
    const int lane = tid & 31;
    const int lo   = lane & 15;
    const int hi   = lane >> 4;
    const int wave = tid >> 5;

    const int rowblk = blockIdx.x;   // 0..63
    const int v      = blockIdx.y;   // 0..47
    const int row0   = rowblk * BT;

    const float* W1v = W1 + (size_t)v * IN_ * H_;
    const float* W2v = W2 + (size_t)v * H_  * H_;
    const float* W3v = W3 + (size_t)v * H_  * OUT_;
    const float* b1v = b1 + v * H_;
    const float* b2v = b2 + v * H_;

    // wave patch for layers 1&2: 32 rows x 64 cols (2x4 tiles of 16x16)
    const int mb = (wave >> 3) * 32;
    const int nb = (wave & 7) * 64;

    v8f   acc[2][4];
    f32x4 wrA[8], wrB[8], xr[2];

    //=======================================================================
    // Layer 1: h1 = relu(x @ W1v + b1v)        K = 1024, 16 chunks of 64
    //=======================================================================
    #pragma unroll
    for (int i = 0; i < 2; ++i)
        #pragma unroll
        for (int j = 0; j < 4; ++j) acc[i][j] = (v8f){};

    // prologue: stage chunk 0 into buffer 0
    load_x_chunk(x, row0, 0, tid, xr);
    load_w_chunk(W1v, tid, wrA, wrB);
    cvt_store_x(&lds[0], tid, xr);
    cvt_store_w(&lds[SW_BASE], tid, wrA, wrB);
    __syncthreads();

    for (int kc = 0; kc < IN_ / 64; ++kc) {
        const int cur = kc & 1;
        const int nxt = cur ^ 1;
        // issue next chunk's global loads first (latency hides under WMMA)
        if (kc + 1 < IN_ / 64) {
            load_x_chunk(x, row0, kc + 1, tid, xr);
            load_w_chunk(W1v + (size_t)(kc + 1) * 64 * H_, tid, wrA, wrB);
        }
        // compute current chunk from LDS
        #pragma unroll
        for (int ks = 0; ks < 2; ++ks)
            mma_step(&lds[cur * SX_BUF], SX_STR, ks * 32,
                     &lds[SW_BASE + cur * SW_BUF], ks * 32,
                     mb, nb, lo, hi, acc);
        // convert + store next chunk into the other buffer
        if (kc + 1 < IN_ / 64) {
            cvt_store_x(&lds[nxt * SX_BUF], tid, xr);
            cvt_store_w(&lds[SW_BASE + nxt * SW_BUF], tid, wrA, wrB);
        }
        __syncthreads();
    }

    // bias + relu -> sH1 (bf16, [64][520])
    #pragma unroll
    for (int j = 0; j < 4; ++j) {
        int n = nb + j * 16 + lo;
        float bias = b1v[n];
        #pragma unroll
        for (int i = 0; i < 2; ++i)
            #pragma unroll
            for (int r = 0; r < 8; ++r) {
                int m = mb + i * 16 + r + hi * 8;
                float val = fmaxf(acc[i][j][r] + bias, 0.0f);
                lds[SH1_OFF + m * SH_STR + n] = (__bf16)val;
            }
    }
    __syncthreads();

    //=======================================================================
    // Layer 2: h2 = relu(h1 @ W2v + b2v)       K = 512, 8 chunks of 64
    //=======================================================================
    #pragma unroll
    for (int i = 0; i < 2; ++i)
        #pragma unroll
        for (int j = 0; j < 4; ++j) acc[i][j] = (v8f){};

    load_w_chunk(W2v, tid, wrA, wrB);
    cvt_store_w(&lds[SW_BASE], tid, wrA, wrB);
    __syncthreads();

    for (int kc = 0; kc < H_ / 64; ++kc) {
        const int cur = kc & 1;
        const int nxt = cur ^ 1;
        if (kc + 1 < H_ / 64)
            load_w_chunk(W2v + (size_t)(kc + 1) * 64 * H_, tid, wrA, wrB);
        #pragma unroll
        for (int ks = 0; ks < 2; ++ks)
            mma_step(&lds[SH1_OFF], SH_STR, kc * 64 + ks * 32,
                     &lds[SW_BASE + cur * SW_BUF], ks * 32,
                     mb, nb, lo, hi, acc);
        if (kc + 1 < H_ / 64)
            cvt_store_w(&lds[SW_BASE + nxt * SW_BUF], tid, wrA, wrB);
        __syncthreads();
    }

    #pragma unroll
    for (int j = 0; j < 4; ++j) {
        int n = nb + j * 16 + lo;
        float bias = b2v[n];
        #pragma unroll
        for (int i = 0; i < 2; ++i)
            #pragma unroll
            for (int r = 0; r < 8; ++r) {
                int m = mb + i * 16 + r + hi * 8;
                float val = fmaxf(acc[i][j][r] + bias, 0.0f);
                lds[SH2_OFF + m * SH_STR + n] = (__bf16)val;
            }
    }
    __syncthreads();

    //=======================================================================
    // Layer 3: o = h2 @ W3v                     K = 512, N = 64
    //=======================================================================
    // stage W3 transposed: lds[n][k] (stride 520) in SW buffer 0
    #pragma unroll
    for (int it = 0; it < 8; ++it) {
        int u = it * NTHR + tid;            // 0..4095 units of (2k x 4n)
        int n = (u & 15) * 4;
        int k = (u >> 4) * 2;
        f32x4 d0 = *(const f32x4*)&W3v[(size_t)k * OUT_ + n];
        f32x4 d1 = *(const f32x4*)&W3v[(size_t)(k + 1) * OUT_ + n];
        #pragma unroll
        for (int j = 0; j < 4; ++j) {
            bf16x2 p;
            p.x = (__bf16)d0[j];
            p.y = (__bf16)d1[j];
            *(bf16x2*)&lds[SW_BASE + (n + j) * SW3_STR + k] = p;
        }
    }
    __syncthreads();

    // 16 output tiles (4x4 of 16x16) over 16 waves -> 1 tile each
    {
        const int tm0 = (wave >> 2) * 16;
        const int tn0 = (wave & 3) * 16;
        v8f c = (v8f){};
        #pragma unroll
        for (int k0 = 0; k0 < H_; k0 += 32) {
            const __bf16* ar = &lds[SH2_OFF + (tm0 + lo) * SH_STR + k0 + hi * 8];
            v16bf A = ld_frag(ar, ar + 16);
            const __bf16* br = &lds[SW_BASE + (tn0 + lo) * SW3_STR + k0 + hi * 8];
            v16bf Bv = ld_frag(br, br + 16);
            c = __builtin_amdgcn_wmma_f32_16x16x32_bf16(
                false, A, false, Bv, (short)0, c, false, false);
        }
        #pragma unroll
        for (int r = 0; r < 8; ++r) {
            int m = tm0 + r + hi * 8;
            out[(size_t)(row0 + m) * (V_ * OUT_) + v * OUT_ + tn0 + lo] = c[r];
        }
    }
}

extern "C" void kernel_launch(void* const* d_in, const int* in_sizes, int n_in,
                              void* d_out, int out_size, void* d_ws, size_t ws_size,
                              hipStream_t stream) {
    (void)in_sizes; (void)n_in; (void)d_ws; (void)ws_size; (void)out_size;
    const float* x  = (const float*)d_in[0];
    const float* W1 = (const float*)d_in[1];
    const float* b1 = (const float*)d_in[2];
    const float* W2 = (const float*)d_in[3];
    const float* b2 = (const float*)d_in[4];
    const float* W3 = (const float*)d_in[5];
    float* out = (float*)d_out;

    dim3 grid(B_ / BT, V_);   // 64 x 48 = 3072 blocks
    mlp48_fused_kernel<<<grid, NTHR, 0, stream>>>(x, W1, b1, W2, b2, W3, out);
}
